// AttentionBlock_7017976561956
// MI455X (gfx1250) — compile-verified, matrix operand order checked by reading, not run on hardware
//
#include <hip/hip_runtime.h>
#include <hip/hip_bf16.h>
#include <stdint.h>

#define D 1024
#define SLEN 4096
#define BATCH 4
#define NROWS (BATCH * SLEN)   // 16384
#define KTILES (D / 32)        // 32

typedef __bf16 bf16_t;
typedef __attribute__((ext_vector_type(16))) __bf16 v16bf;
typedef __attribute__((ext_vector_type(8)))  float  v8f;

union AFrag { uint32_t u[8]; v16bf v; };

#if defined(__gfx1250__) && __has_builtin(__builtin_amdgcn_global_load_async_to_lds_b128)
#define USE_ASYNC_COPY 1
// exact parameter type per hipcc diagnostic: native int vector (16B), generic AS
typedef int v4i_vec __attribute__((__vector_size__(16)));
#else
#define USE_ASYNC_COPY 0
#endif

#if defined(__gfx1250__) && __has_builtin(__builtin_amdgcn_s_wait_asynccnt)
#define ASYNC_WAIT(n) __builtin_amdgcn_s_wait_asynccnt((short)(n))
#else
#define ASYNC_WAIT(n) asm volatile("s_wait_asynccnt %0" :: "i"(n) : "memory")
#endif

// ---------------------------------------------------------------------------
// Stage 1 of column sum: each block reduces 64 rows of one batch into a
// partial-sum slot (deterministic: fixed summation order).
__global__ void colsum_part_kernel(const float* __restrict__ x,
                                   float* __restrict__ part) {
    int b  = blockIdx.x >> 6;        // 4 batches
    int ch = blockIdx.x & 63;        // 64 chunks of 64 rows
    const float* xb = x + (size_t)b * SLEN * D + (size_t)ch * 64 * D;
    for (int c = threadIdx.x; c < D; c += 256) {
        float s = 0.f;
        #pragma unroll 4
        for (int r = 0; r < 64; ++r) s += xb[(size_t)r * D + c];
        part[((size_t)(b * 64 + ch)) * D + c] = s;
    }
}

// Stage 2: reduce the 64 partials per batch into xsum (deterministic).
__global__ void colsum_final_kernel(const float* __restrict__ part,
                                    float* __restrict__ xsum) {
    int b = blockIdx.x;
    for (int c = threadIdx.x; c < D; c += 256) {
        float s = 0.f;
        for (int ch = 0; ch < 64; ++ch)
            s += part[((size_t)(b * 64 + ch)) * D + c];
        xsum[b * D + c] = s;
    }
}

// ---------------------------------------------------------------------------
// out[b][e] = dot(W[e,:], vin[b,:])   (one wave32 per output element)
__global__ void matvec_kernel(const float* __restrict__ W,
                              const float* __restrict__ vin,
                              float* __restrict__ vout) {
    int gw   = blockIdx.x * 8 + (threadIdx.x >> 5);   // global wave id
    int b    = gw >> 10;                               // /1024
    int e    = gw & (D - 1);
    int lane = threadIdx.x & 31;
    const float* wr = W + (size_t)e * D;
    const float* vv = vin + (size_t)b * D;
    float s = 0.f;
    #pragma unroll 8
    for (int k = lane; k < D; k += 32) s += wr[k] * vv[k];
    #pragma unroll
    for (int o = 16; o > 0; o >>= 1) s += __shfl_down(s, o, 32);
    if (lane == 0) vout[b * D + e] = s;
}

// ---------------------------------------------------------------------------
// combined block reduction of (sum, sumsq) over 256 threads (8 waves)
__device__ inline void block_reduce2(float& s, float& s2, float* sh) {
    #pragma unroll
    for (int o = 16; o > 0; o >>= 1) {
        s  += __shfl_down(s,  o, 32);
        s2 += __shfl_down(s2, o, 32);
    }
    int lane = threadIdx.x & 31, w = threadIdx.x >> 5;
    if (lane == 0) { sh[w] = s; sh[8 + w] = s2; }
    __syncthreads();
    if (w == 0) {
        float a  = (lane < 8) ? sh[lane]     : 0.f;
        float b2 = (lane < 8) ? sh[8 + lane] : 0.f;
        #pragma unroll
        for (int o = 4; o > 0; o >>= 1) {
            a  += __shfl_down(a,  o, 32);
            b2 += __shfl_down(b2, o, 32);
        }
        if (lane == 0) { sh[0] = a; sh[8] = b2; }
    }
    __syncthreads();
    s = sh[0]; s2 = sh[8];
    __syncthreads();
}

// x1[n,:] = LN(x[n,:] + avec[b,:]) * g + beta; writes f32 + bf16 copies.
__global__ void ln1_kernel(const float* __restrict__ x,
                           const float* __restrict__ avec,
                           const float* __restrict__ g,
                           const float* __restrict__ beta,
                           float* __restrict__ x1,
                           bf16_t* __restrict__ x1bf) {
    __shared__ float sh[16];
    int n = blockIdx.x;
    int b = n >> 12;                 // n / 4096
    const float* xr = x + (size_t)n * D;
    const float* ar = avec + (size_t)b * D;
    float y[4], s = 0.f, s2 = 0.f;
    #pragma unroll
    for (int i = 0; i < 4; ++i) {
        int d = threadIdx.x + i * 256;
        float v = xr[d] + ar[d];
        y[i] = v; s += v; s2 += v * v;
    }
    block_reduce2(s, s2, sh);
    float mean = s * (1.f / D);
    float var  = s2 * (1.f / D) - mean * mean;
    float rstd = rsqrtf(var + 1e-5f);
    #pragma unroll
    for (int i = 0; i < 4; ++i) {
        int d = threadIdx.x + i * 256;
        float o = (y[i] - mean) * rstd * g[d] + beta[d];
        x1[(size_t)n * D + d]   = o;
        x1bf[(size_t)n * D + d] = (bf16_t)o;
    }
}

// io (= mlp_out + b2 in d_out) : io[n,:] = LN(x1[n,:] + io[n,:]) * g + beta
__global__ void ln2_kernel(const float* __restrict__ x1,
                           float* __restrict__ io,
                           const float* __restrict__ g,
                           const float* __restrict__ beta) {
    __shared__ float sh[16];
    int n = blockIdx.x;
    const float* xr = x1 + (size_t)n * D;
    float* ior = io + (size_t)n * D;
    float y[4], s = 0.f, s2 = 0.f;
    #pragma unroll
    for (int i = 0; i < 4; ++i) {
        int d = threadIdx.x + i * 256;
        float v = xr[d] + ior[d];
        y[i] = v; s += v; s2 += v * v;
    }
    block_reduce2(s, s2, sh);
    float mean = s * (1.f / D);
    float var  = s2 * (1.f / D) - mean * mean;
    float rstd = rsqrtf(var + 1e-5f);
    #pragma unroll
    for (int i = 0; i < 4; ++i) {
        int d = threadIdx.x + i * 256;
        ior[d] = (y[i] - mean) * rstd * g[d] + beta[d];
    }
}

// ---------------------------------------------------------------------------
__global__ void cvt_bf16_kernel(const float* __restrict__ src,
                                bf16_t* __restrict__ dst, int n) {
    int i = blockIdx.x * 256 + threadIdx.x;
    if (i < n) dst[i] = (bf16_t)src[i];
}

// ---------------------------------------------------------------------------
// GEMM: out[n,e] = epilogue( sum_k A[n,k] * W[e,k] + bias[e] ), K = E = 1024.
//   A: bf16 row-major [NROWS,1024]; W: bf16 row-major [1024,1024].
// Block tile 128x128, 8 waves each computing 64x32 (4x2 WMMA C tiles).
// Tiles staged through double-buffered LDS; on gfx1250 the copies use
// GLOBAL_LOAD_ASYNC_TO_LDS_B128 (ASYNCcnt) overlapped with the WMMAs.
// LDS row stride = 20 dwords (80B): 16B-aligned chunks, conflict-free gathers.
__global__ void __launch_bounds__(256)
gemm_bias_kernel(const bf16_t* __restrict__ A,
                 const bf16_t* __restrict__ W,
                 const float* __restrict__ bias,
                 bf16_t* __restrict__ out_bf,
                 float* __restrict__ out_f32,
                 int gelu_flag) {
    __shared__ uint32_t aLds[2][128 * 20];
    __shared__ uint32_t bLds[2][128 * 20];

    const int bm = blockIdx.x >> 3;        // 128 row blocks
    const int bn = blockIdx.x & 7;         // 8 col blocks
    const int tid  = threadIdx.x;
    const int lane = tid & 31;
    const int wave = tid >> 5;
    const int wm = wave >> 2;              // 0..1
    const int wn = wave & 3;               // 0..3
    const int frow = lane & 15;            // M/N index within 16
    const int kq   = (lane >> 4) << 2;     // dword offset of K-half (0 or 4)

    v8f acc[4][2];
    #pragma unroll
    for (int i = 0; i < 4; ++i)
        #pragma unroll
        for (int j = 0; j < 2; ++j)
            acc[i][j] = (v8f){0.f,0.f,0.f,0.f,0.f,0.f,0.f,0.f};

    // 128x32 bf16 tile = 512 chunks of 16B; each thread owns 2 chunks.
    // chunk c = tid + it*256 ; row = c>>2 ; q = c&3 (16B sub-chunk in row)
    auto compute = [&](int buf) {
        AFrag fa[4], fb[2];
        #pragma unroll
        for (int i = 0; i < 4; ++i) {
            const uint32_t* base = &aLds[buf][(wm * 64 + i * 16 + frow) * 20 + kq];
            #pragma unroll
            for (int j = 0; j < 4; ++j) { fa[i].u[j] = base[j]; fa[i].u[4 + j] = base[8 + j]; }
        }
        #pragma unroll
        for (int j = 0; j < 2; ++j) {
            const uint32_t* base = &bLds[buf][(wn * 32 + j * 16 + frow) * 20 + kq];
            #pragma unroll
            for (int t = 0; t < 4; ++t) { fb[j].u[t] = base[t]; fb[j].u[4 + t] = base[8 + t]; }
        }
        #pragma unroll
        for (int i = 0; i < 4; ++i)
            #pragma unroll
            for (int j = 0; j < 2; ++j)
                acc[i][j] = __builtin_amdgcn_wmma_f32_16x16x32_bf16(
                    false, fa[i].v, false, fb[j].v,
                    (short)0, acc[i][j], false, false);
    };

#if USE_ASYNC_COPY
    auto issue_copy = [&](int buf, int k0) {
        #pragma unroll
        for (int it = 0; it < 2; ++it) {
            int c = tid + it * 256;
            int row = c >> 2, q = c & 3;
            const bf16_t* ga = &A[(size_t)(bm * 128 + row) * D + k0 + q * 8];
            const bf16_t* gb = &W[(size_t)(bn * 128 + row) * D + k0 + q * 8];
            char* la = (char*)&aLds[buf][0] + row * 80 + q * 16;
            char* lb = (char*)&bLds[buf][0] + row * 80 + q * 16;
            __builtin_amdgcn_global_load_async_to_lds_b128(
                (v4i_vec*)ga, (v4i_vec*)la, 0, 0);
            __builtin_amdgcn_global_load_async_to_lds_b128(
                (v4i_vec*)gb, (v4i_vec*)lb, 0, 0);
        }
    };

    issue_copy(0, 0);
    for (int kt = 0; kt < KTILES; ++kt) {
        int buf = kt & 1;
        if (kt + 1 < KTILES) {
            issue_copy(buf ^ 1, (kt + 1) * 32);
            ASYNC_WAIT(4);      // in-order: the 4 ops for buf are complete
        } else {
            ASYNC_WAIT(0);
        }
        __syncthreads();        // all waves' copies for buf complete
        compute(buf);
        __syncthreads();        // all waves done reading buf
    }
#else
    uint4 ra[2], rb[2];
    auto stage_load = [&](int k0) {
        #pragma unroll
        for (int it = 0; it < 2; ++it) {
            int c = tid + it * 256;
            int row = c >> 2, q = c & 3;
            ra[it] = *(const uint4*)&A[(size_t)(bm * 128 + row) * D + k0 + q * 8];
            rb[it] = *(const uint4*)&W[(size_t)(bn * 128 + row) * D + k0 + q * 8];
        }
    };
    auto store_staged = [&](int buf) {
        #pragma unroll
        for (int it = 0; it < 2; ++it) {
            int c = tid + it * 256;
            int row = c >> 2, q = c & 3;
            *(uint4*)((char*)&aLds[buf][0] + row * 80 + q * 16) = ra[it];
            *(uint4*)((char*)&bLds[buf][0] + row * 80 + q * 16) = rb[it];
        }
    };

    stage_load(0);
    for (int kt = 0; kt < KTILES; ++kt) {
        int buf = kt & 1;
        store_staged(buf);
        if (kt + 1 < KTILES) stage_load((kt + 1) * 32);  // overlaps compute
        __syncthreads();
        compute(buf);
        __syncthreads();
    }
#endif

    // --- epilogue: C layout -> lane holds column (lane&15), rows r + 8*(lane>>4)
    const int ccol   = lane & 15;
    const int rowoff = (lane >> 4) << 3;
    #pragma unroll
    for (int i = 0; i < 4; ++i) {
        #pragma unroll
        for (int j = 0; j < 2; ++j) {
            int gm = bm * 128 + wm * 64 + i * 16 + rowoff;
            int gn = bn * 128 + wn * 32 + j * 16 + ccol;
            float bv = bias[gn];
            #pragma unroll
            for (int r = 0; r < 8; ++r) {
                float v = acc[i][j][r] + bv;
                if (gelu_flag) {
                    v = 0.5f * v * (1.0f + erff(v * 0.70710678118654752f));
                    out_bf[(size_t)(gm + r) * D + gn] = (bf16_t)v;
                } else {
                    out_f32[(size_t)(gm + r) * D + gn] = v;
                }
            }
        }
    }
}

// ---------------------------------------------------------------------------
extern "C" void kernel_launch(void* const* d_in, const int* in_sizes, int n_in,
                              void* d_out, int out_size, void* d_ws, size_t ws_size,
                              hipStream_t stream) {
    (void)in_sizes; (void)n_in; (void)out_size; (void)ws_size;
    const float* x     = (const float*)d_in[0];
    const float* w_qkv = (const float*)d_in[1];
    const float* w_o   = (const float*)d_in[2];
    const float* w1    = (const float*)d_in[3];
    const float* b1    = (const float*)d_in[4];
    const float* w2    = (const float*)d_in[5];
    const float* b2    = (const float*)d_in[6];
    const float* ln1g  = (const float*)d_in[7];
    const float* ln1b  = (const float*)d_in[8];
    const float* ln2g  = (const float*)d_in[9];
    const float* ln2b  = (const float*)d_in[10];
    float* out = (float*)d_out;

    char* ws = (char*)d_ws;
    const size_t KB = 1024, MB = 1024 * 1024;
    float*  xsum = (float*)(ws + 0);
    float*  vsum = (float*)(ws + 16 * KB);
    float*  avec = (float*)(ws + 32 * KB);
    float*  part = (float*)(ws + 48 * KB);                  // 1 MB
    bf16_t* w1bf = (bf16_t*)(ws + 48 * KB + 1 * MB);        // 2 MB
    bf16_t* w2bf = (bf16_t*)(ws + 48 * KB + 3 * MB);        // 2 MB
    float*  x1   = (float*)(ws + 48 * KB + 5 * MB);         // 64 MB
    bf16_t* x1bf = (bf16_t*)(ws + 48 * KB + 69 * MB);       // 32 MB
    bf16_t* h    = (bf16_t*)(ws + 48 * KB + 101 * MB);      // 32 MB

    // attention collapses to two matvecs on the batch column-sum of x
    colsum_part_kernel <<<BATCH * 64, 256, 0, stream>>>(x, part);
    colsum_final_kernel<<<BATCH,      256, 0, stream>>>(part, xsum);
    matvec_kernel<<<BATCH * D / 8, 256, 0, stream>>>(w_qkv + 2 * (size_t)D * D, xsum, vsum);
    matvec_kernel<<<BATCH * D / 8, 256, 0, stream>>>(w_o, vsum, avec);

    ln1_kernel<<<NROWS, 256, 0, stream>>>(x, avec, ln1g, ln1b, x1, x1bf);

    cvt_bf16_kernel<<<(D * D + 255) / 256, 256, 0, stream>>>(w1, w1bf, D * D);
    cvt_bf16_kernel<<<(D * D + 255) / 256, 256, 0, stream>>>(w2, w2bf, D * D);

    dim3 ggrid((NROWS / 128) * (D / 128));   // 1024 blocks
    // h = gelu(x1 @ w1^T + b1)  -> bf16
    gemm_bias_kernel<<<ggrid, 256, 0, stream>>>(x1bf, w1bf, b1, h, nullptr, 1);
    // d_out = h @ w2^T + b2     -> f32
    gemm_bias_kernel<<<ggrid, 256, 0, stream>>>(h, w2bf, b2, nullptr, out, 0);

    // final LN in-place on d_out with residual x1
    ln2_kernel<<<NROWS, 256, 0, stream>>>(x1, out, ln2g, ln2b);
}
